// Decoder_35905926595425
// MI455X (gfx1250) — matrix operand order repaired and checked by reading, not compiled, not run on hardware
//
#include <hip/hip_runtime.h>
#include <math.h>

// Problem dims (fixed by the reference)
#define S_DIM 196
#define B_DIM 64
#define E_DIM 1024
#define D_DIM 1800
#define W_DIM 512
#define V_DIM 32000

typedef __attribute__((ext_vector_type(16))) __bf16 v16bf;
typedef __attribute__((ext_vector_type(8)))  float  v8f;

#define FLAG_BIAS 1
#define FLAG_RELU 2
#define FLAG_ADDC 4

__device__ __forceinline__ float sigmoid_(float x) { return 1.0f / (1.0f + expf(-x)); }

// Convert 8 consecutive fp32 (16B-aligned) into bf16 fragment elements [base..base+7]
__device__ __forceinline__ void cvt8(v16bf& o, int base, const float* p) {
  const float4* q = (const float4*)p;
  float4 a = q[0], b = q[1];
  o[base + 0] = (__bf16)a.x; o[base + 1] = (__bf16)a.y;
  o[base + 2] = (__bf16)a.z; o[base + 3] = (__bf16)a.w;
  o[base + 4] = (__bf16)b.x; o[base + 5] = (__bf16)b.y;
  o[base + 6] = (__bf16)b.z; o[base + 7] = (__bf16)b.w;
}

// K-tail fragment loads: branch-free (clamped address + select-to-zero).
// A layout: lane = 16*half + (m-m0); elems 0..7 -> k = k0+8*half+e ; 8..15 -> +16
__device__ __forceinline__ v16bf tail_A(const float* row, int half, int k0, int K) {
  v16bf o;
#pragma unroll
  for (int e = 0; e < 8; ++e) {
    int ka = k0 + half * 8 + e;
    int kb = ka + 16;
    float fa = row[ka < K ? ka : K - 1];
    float fb = row[kb < K ? kb : K - 1];
    o[e]     = (__bf16)(ka < K ? fa : 0.0f);
    o[8 + e] = (__bf16)(kb < K ? fb : 0.0f);
  }
  return o;
}
// B layout: lane = 16*half + (n-n0); elems 0..7 -> k = k0+8*half+e ; 8..15 -> +16
__device__ __forceinline__ v16bf tail_B(const float* row, int half, int k0, int K) {
  v16bf o;
#pragma unroll
  for (int e = 0; e < 16; ++e) {
    int k = k0 + half * 8 + (e < 8 ? e : e + 8);
    float f = row[k < K ? k : K - 1];
    o[e] = (__bf16)(k < K ? f : 0.0f);
  }
  return o;
}

#define WMMA_BF16(A_, B_, C_) \
  __builtin_amdgcn_wmma_f32_16x16x32_bf16(false, (A_), false, (B_), (short)0, (C_), false, false)

// Generic  C[M,N] = act( A[M,K] * W[N,K]^T + bias + Cadd )
// Requires: M multiple of 64 and fully covered by the grid (true for all call sites).
// Block: 128 threads = 4 waves; block tile 64M x 128N; wave tile 32M x 64N.
__global__ __launch_bounds__(128) void gemm_bias_act(
    const float* __restrict__ A, const float* __restrict__ W,
    const float* __restrict__ bias, const float* __restrict__ Cadd,
    float* __restrict__ Cout,
    int N, int K, int lda, int ldw, int ldc, int flags) {
  const int lane = threadIdx.x & 31;
  const int wave = threadIdx.x >> 5;
  const int half = lane >> 4;
  const int l16  = lane & 15;
  const int m0 = blockIdx.y * 64 + (wave & 1) * 32;
  const int n0 = blockIdx.x * 128 + (wave >> 1) * 64;

  // Per-lane base pointers (hoisted out of the k-loop). B rows clamped to N-1:
  // out-of-range columns hold garbage that is never stored.
  const float* aRow0 = A + (size_t)(m0 + l16) * lda;
  const float* aRow1 = A + (size_t)(m0 + 16 + l16) * lda;
  const float* aB0 = aRow0 + half * 8;
  const float* aB1 = aRow1 + half * 8;
  const float* bRow[4];
  const float* bB[4];
#pragma unroll
  for (int j = 0; j < 4; ++j) {
    int n = n0 + 16 * j + l16; if (n > N - 1) n = N - 1;
    bRow[j] = W + (size_t)n * ldw;
    bB[j]   = bRow[j] + half * 8;
  }

  v8f zero = {0.f, 0.f, 0.f, 0.f, 0.f, 0.f, 0.f, 0.f};
  v8f acc[2][4];
#pragma unroll
  for (int i = 0; i < 2; ++i)
#pragma unroll
    for (int j = 0; j < 4; ++j) acc[i][j] = zero;

  const int Kfull = K & ~31;
  for (int k0 = 0; k0 < Kfull; k0 += 32) {
    v16bf a0, a1;
    cvt8(a0, 0, aB0 + k0); cvt8(a0, 8, aB0 + k0 + 16);
    cvt8(a1, 0, aB1 + k0); cvt8(a1, 8, aB1 + k0 + 16);
#pragma unroll
    for (int j = 0; j < 4; ++j) {
      __builtin_prefetch(bB[j] + k0 + 64, 0, 1);  // speculative stream-ahead on weights
      v16bf b;
      cvt8(b, 0, bB[j] + k0); cvt8(b, 8, bB[j] + k0 + 16);
      acc[0][j] = WMMA_BF16(a0, b, acc[0][j]);
      acc[1][j] = WMMA_BF16(a1, b, acc[1][j]);
    }
  }
  if (Kfull < K) {  // uniform tail step, branch-free lanes
    v16bf a0 = tail_A(aRow0, half, Kfull, K);
    v16bf a1 = tail_A(aRow1, half, Kfull, K);
#pragma unroll
    for (int j = 0; j < 4; ++j) {
      v16bf b = tail_B(bRow[j], half, Kfull, K);
      acc[0][j] = WMMA_BF16(a0, b, acc[0][j]);
      acc[1][j] = WMMA_BF16(a1, b, acc[1][j]);
    }
  }

#pragma unroll
  for (int i = 0; i < 2; ++i) {
#pragma unroll
    for (int j = 0; j < 4; ++j) {
      int n = n0 + 16 * j + l16;
      if (n < N) {
        float bv = (flags & FLAG_BIAS) ? bias[n] : 0.0f;
#pragma unroll
        for (int r = 0; r < 8; ++r) {
          int m = m0 + 16 * i + half * 8 + r;
          float v = acc[i][j][r] + bv;
          if (flags & FLAG_ADDC) v += Cadd[(size_t)m * ldc + n];
          if (flags & FLAG_RELU) v = fmaxf(v, 0.0f);
          Cout[(size_t)m * ldc + n] = v;
        }
      }
    }
  }
}

// Fused attention scores: for m = s*B + b:
//   u[m,n] = (H[m,:] @ Wa1h[n,:]) + qproj[b,n];  partial[m,jx] = sum_n relu(u) * wa2[n]
// One wave per block; wave tile 32M x 64N. Deterministic per-block partials.
__global__ __launch_bounds__(32) void attn_scores_partial(
    const float* __restrict__ H,     // [S*B, K]
    const float* __restrict__ Wa1h,  // leading dim ldw (= D+E), h-part columns
    const float* __restrict__ qproj, // [B, N] (includes ba1)
    const float* __restrict__ wa2,   // [N]
    float* __restrict__ partial,     // [M, gridDim.x]
    int N, int K, int ldw) {
  const int lane = threadIdx.x & 31;
  const int half = lane >> 4;
  const int l16  = lane & 15;
  const int m0 = blockIdx.y * 32;
  const int n0 = blockIdx.x * 64;

  const float* aB0 = H + (size_t)(m0 + l16) * K + half * 8;
  const float* aB1 = H + (size_t)(m0 + 16 + l16) * K + half * 8;
  const float* bB[4];
#pragma unroll
  for (int j = 0; j < 4; ++j) {
    int n = n0 + 16 * j + l16; if (n > N - 1) n = N - 1;
    bB[j] = Wa1h + (size_t)n * ldw + half * 8;
  }

  v8f zero = {0.f, 0.f, 0.f, 0.f, 0.f, 0.f, 0.f, 0.f};
  v8f acc[2][4];
#pragma unroll
  for (int i = 0; i < 2; ++i)
#pragma unroll
    for (int j = 0; j < 4; ++j) acc[i][j] = zero;

  for (int k0 = 0; k0 < K; k0 += 32) {  // K = E = 1024, multiple of 32
    v16bf a0, a1;
    cvt8(a0, 0, aB0 + k0); cvt8(a0, 8, aB0 + k0 + 16);
    cvt8(a1, 0, aB1 + k0); cvt8(a1, 8, aB1 + k0 + 16);
#pragma unroll
    for (int j = 0; j < 4; ++j) {
      __builtin_prefetch(bB[j] + k0 + 64, 0, 1);
      v16bf b;
      cvt8(b, 0, bB[j] + k0); cvt8(b, 8, bB[j] + k0 + 16);
      acc[0][j] = WMMA_BF16(a0, b, acc[0][j]);
      acc[1][j] = WMMA_BF16(a1, b, acc[1][j]);
    }
  }

  float rsum[2][8];
#pragma unroll
  for (int i = 0; i < 2; ++i)
#pragma unroll
    for (int r = 0; r < 8; ++r) rsum[i][r] = 0.0f;

#pragma unroll
  for (int i = 0; i < 2; ++i)
#pragma unroll
    for (int j = 0; j < 4; ++j) {
      int n  = n0 + 16 * j + l16;
      int na = n < N ? n : N - 1;            // clamped address
      float w2 = wa2[na];
      if (n >= N) w2 = 0.0f;                 // zero weight kills out-of-range columns
#pragma unroll
      for (int r = 0; r < 8; ++r) {
        int m = m0 + 16 * i + half * 8 + r;
        int b = m & (B_DIM - 1);
        float q = qproj[(size_t)b * N + na];
        float u = fmaxf(acc[i][j][r] + q, 0.0f);
        rsum[i][r] += u * w2;
      }
    }

  // reduce over the 16 lanes of each half-wave (each half holds distinct m rows)
#pragma unroll
  for (int off = 1; off < 16; off <<= 1)
#pragma unroll
    for (int i = 0; i < 2; ++i)
#pragma unroll
      for (int r = 0; r < 8; ++r)
        rsum[i][r] += __shfl_xor(rsum[i][r], off, 32);

  if (l16 == 0) {
#pragma unroll
    for (int i = 0; i < 2; ++i)
#pragma unroll
      for (int r = 0; r < 8; ++r) {
        int m = m0 + 16 * i + half * 8 + r;
        partial[(size_t)m * gridDim.x + blockIdx.x] = rsum[i][r];
      }
  }
}

__global__ void reduce_scores(const float* __restrict__ partial, float* __restrict__ scores,
                              const float* __restrict__ ba2, int M, int J) {
  int m = blockIdx.x * blockDim.x + threadIdx.x;
  if (m >= M) return;
  float s = ba2[0];
  for (int j = 0; j < J; ++j) s += partial[(size_t)m * J + j];
  scores[m] = s;
}

__global__ __launch_bounds__(256) void softmax_over_s(const float* __restrict__ scores,
                                                      float* __restrict__ aw) {
  int b = blockIdx.x, t = threadIdx.x;
  __shared__ float red[256];
  float mx = -3.402823466e38f;
  for (int s = t; s < S_DIM; s += 256) mx = fmaxf(mx, scores[s * B_DIM + b]);
  red[t] = mx; __syncthreads();
  for (int w = 128; w > 0; w >>= 1) { if (t < w) red[t] = fmaxf(red[t], red[t + w]); __syncthreads(); }
  mx = red[0]; __syncthreads();
  float sum = 0.0f;
  for (int s = t; s < S_DIM; s += 256) sum += expf(scores[s * B_DIM + b] - mx);
  red[t] = sum; __syncthreads();
  for (int w = 128; w > 0; w >>= 1) { if (t < w) red[t] += red[t + w]; __syncthreads(); }
  float inv = 1.0f / red[0];
  for (int s = t; s < S_DIM; s += 256)
    aw[s * B_DIM + b] = expf(scores[s * B_DIM + b] - mx) * inv;
}

__global__ __launch_bounds__(256) void beta_kernel(const float* __restrict__ h0,
                                                   const float* __restrict__ Wb,
                                                   const float* __restrict__ bb,
                                                   float* __restrict__ beta) {
  int b = blockIdx.x, t = threadIdx.x;
  __shared__ float red[256];
  float s = 0.0f;
  for (int d = t; d < D_DIM; d += 256) s += h0[(size_t)b * D_DIM + d] * Wb[d];
  red[t] = s; __syncthreads();
  for (int w = 128; w > 0; w >>= 1) { if (t < w) red[t] += red[t + w]; __syncthreads(); }
  if (t == 0) beta[b] = sigmoid_(red[0] + bb[0]);
}

__global__ void mean_over_s(const float* __restrict__ H, float* __restrict__ havg) {
  int t = blockIdx.x * blockDim.x + threadIdx.x;
  if (t >= B_DIM * E_DIM) return;
  float s = 0.0f;
  for (int si = 0; si < S_DIM; ++si) s += H[(size_t)si * B_DIM * E_DIM + t];
  havg[t] = s * (1.0f / S_DIM);
}

__global__ void ctx_kernel(const float* __restrict__ aw, const float* __restrict__ H,
                           const float* __restrict__ beta, float* __restrict__ ctx) {
  int t = blockIdx.x * blockDim.x + threadIdx.x;
  if (t >= B_DIM * E_DIM) return;
  int b = t >> 10;
  int e = t & (E_DIM - 1);
  float s = 0.0f;
  for (int si = 0; si < S_DIM; ++si)
    s += aw[si * B_DIM + b] * H[((size_t)si * B_DIM + b) * E_DIM + e];
  ctx[t] = s * beta[b];
}

__global__ void build_x(const int* __restrict__ E_tm1, const float* __restrict__ emb,
                        const float* __restrict__ ctx, float* __restrict__ x) {
  const int L = W_DIM + E_DIM;  // 1536
  int t = blockIdx.x * blockDim.x + threadIdx.x;
  if (t >= B_DIM * L) return;
  int b = t / L, j = t - b * L;
  x[t] = (j < W_DIM) ? emb[(size_t)E_tm1[b] * W_DIM + j] : ctx[(size_t)b * E_DIM + (j - W_DIM)];
}

__global__ void lstm_pointwise(const float* __restrict__ gates, const float* __restrict__ c0,
                               float* __restrict__ h_t, float* __restrict__ c_t) {
  int t = blockIdx.x * blockDim.x + threadIdx.x;
  if (t >= B_DIM * D_DIM) return;
  int b = t / D_DIM, d = t - b * D_DIM;
  const float* g = gates + (size_t)b * 4 * D_DIM;
  float ig = sigmoid_(g[d]);
  float fg = sigmoid_(g[D_DIM + d]);
  float gg = tanhf(g[2 * D_DIM + d]);
  float og = sigmoid_(g[3 * D_DIM + d]);
  float c = fg * c0[t] + ig * gg;
  c_t[t] = c;
  h_t[t] = og * tanhf(c);
}

__global__ __launch_bounds__(256) void argmax_rows(const float* __restrict__ y, int* __restrict__ idx) {
  int b = blockIdx.x, t = threadIdx.x;
  __shared__ float rv[256];
  __shared__ int   ri[256];
  float best = -3.402823466e38f; int bi = V_DIM;
  for (int v = t; v < V_DIM; v += 256) {
    float val = y[(size_t)b * V_DIM + v];
    if (val > best) { best = val; bi = v; }
  }
  rv[t] = best; ri[t] = bi; __syncthreads();
  for (int w = 128; w > 0; w >>= 1) {
    if (t < w) {
      if (rv[t + w] > rv[t] || (rv[t + w] == rv[t] && ri[t + w] < ri[t])) {
        rv[t] = rv[t + w]; ri[t] = ri[t + w];
      }
    }
    __syncthreads();
  }
  if (t == 0) idx[b] = ri[0];
}

__global__ void build_z(const float* __restrict__ h_t, const float* __restrict__ emb,
                        const int* __restrict__ idx, const float* __restrict__ ctx,
                        float* __restrict__ z) {
  const int L = D_DIM + W_DIM + E_DIM;  // 3336
  int t = blockIdx.x * blockDim.x + threadIdx.x;
  if (t >= B_DIM * L) return;
  int b = t / L, j = t - b * L;
  float v;
  if (j < D_DIM)              v = h_t[(size_t)b * D_DIM + j];
  else if (j < D_DIM + W_DIM) v = emb[(size_t)idx[b] * W_DIM + (j - D_DIM)];
  else                        v = ctx[(size_t)b * E_DIM + (j - D_DIM - W_DIM)];
  z[t] = v;
}

static inline dim3 gemm_grid(int M, int N) { return dim3((N + 127) / 128, (M + 63) / 64); }

extern "C" void kernel_launch(void* const* d_in, const int* in_sizes, int n_in,
                              void* d_out, int out_size, void* d_ws, size_t ws_size,
                              hipStream_t stream) {
  (void)in_sizes; (void)n_in; (void)out_size; (void)ws_size;
  const int S = S_DIM, B = B_DIM, E = E_DIM, D = D_DIM, Wd = W_DIM, V = V_DIM;

  const int*   E_tm1 = (const int*)d_in[0];
  const float* y_tm1 = (const float*)d_in[1];
  const float* H     = (const float*)d_in[2];
  const float* emb   = (const float*)d_in[3];
  const float* W_ih  = (const float*)d_in[4];
  const float* b_ih  = (const float*)d_in[5];
  const float* W_hh  = (const float*)d_in[6];
  const float* b_hh  = (const float*)d_in[7];
  const float* W_out = (const float*)d_in[8];
  const float* b_out = (const float*)d_in[9];
  const float* Wh1 = (const float*)d_in[10];
  const float* bh1 = (const float*)d_in[11];
  const float* Wh2 = (const float*)d_in[12];
  const float* bh2 = (const float*)d_in[13];
  const float* Wc1 = (const float*)d_in[14];
  const float* bc1 = (const float*)d_in[15];
  const float* Wc2 = (const float*)d_in[16];
  const float* bc2 = (const float*)d_in[17];
  const float* Wa1 = (const float*)d_in[18];
  const float* ba1 = (const float*)d_in[19];
  const float* Wa2 = (const float*)d_in[20];
  const float* ba2 = (const float*)d_in[21];
  const float* Wb  = (const float*)d_in[22];
  const float* bb  = (const float*)d_in[23];

  // Output layout: logits [B,V], h_t [B,D], c_t [B,D], aw [S,B]
  float* out    = (float*)d_out;
  float* logits = out;
  float* h_t    = logits + (size_t)B * V;
  float* c_t    = h_t + (size_t)B * D;
  float* aw     = c_t + (size_t)B * D;

  // Workspace carve-up
  const int M_attn = S * B;          // 12544
  const int J_attn = (D + 63) / 64;  // 29
  float* ws     = (float*)d_ws;
  float* havg   = ws;                 ws += (size_t)B * E;
  float* t1     = ws;                 ws += (size_t)B * D;
  float* h0     = ws;                 ws += (size_t)B * D;
  float* c0     = ws;                 ws += (size_t)B * D;
  float* qproj  = ws;                 ws += (size_t)B * D;
  float* part   = ws;                 ws += (size_t)M_attn * J_attn;
  float* scores = ws;                 ws += (size_t)M_attn;
  float* beta   = ws;                 ws += 64;
  float* ctx    = ws;                 ws += (size_t)B * E;
  float* x      = ws;                 ws += (size_t)B * (Wd + E);
  float* gates  = ws;                 ws += (size_t)B * 4 * D;
  float* z      = ws;                 ws += (size_t)B * (D + Wd + E);
  int*   amx    = (int*)ws;

  // 1) h_avg = mean_s(h)
  mean_over_s<<<(B * E + 255) / 256, 256, 0, stream>>>(H, havg);

  // 2) h0 = relu(h_avg@Wh1^T+bh1) @ Wh2^T + bh2
  gemm_bias_act<<<gemm_grid(B, D), 128, 0, stream>>>(havg, Wh1, bh1, nullptr, t1,
      D, E, E, E, D, FLAG_BIAS | FLAG_RELU);
  gemm_bias_act<<<gemm_grid(B, D), 128, 0, stream>>>(t1, Wh2, bh2, nullptr, h0,
      D, D, D, D, D, FLAG_BIAS);

  // 3) c0 path (reuse t1)
  gemm_bias_act<<<gemm_grid(B, D), 128, 0, stream>>>(havg, Wc1, bc1, nullptr, t1,
      D, E, E, E, D, FLAG_BIAS | FLAG_RELU);
  gemm_bias_act<<<gemm_grid(B, D), 128, 0, stream>>>(t1, Wc2, bc2, nullptr, c0,
      D, D, D, D, D, FLAG_BIAS);

  // 4) qproj = h0 @ Wa1[:, :D]^T + ba1
  gemm_bias_act<<<gemm_grid(B, D), 128, 0, stream>>>(h0, Wa1, ba1, nullptr, qproj,
      D, D, D, D + E, D, FLAG_BIAS);

  // 5) fused attention scores
  {
    dim3 g(J_attn, M_attn / 32);
    attn_scores_partial<<<g, 32, 0, stream>>>(H, Wa1 + D, qproj, Wa2, part, D, E, D + E);
  }
  reduce_scores<<<(M_attn + 255) / 256, 256, 0, stream>>>(part, scores, ba2, M_attn, J_attn);

  // 6) softmax over S -> aw (output #4)
  softmax_over_s<<<B, 256, 0, stream>>>(scores, aw);

  // 7) beta gate; ctx = beta * sum_s aw * h
  beta_kernel<<<B, 256, 0, stream>>>(h0, Wb, bb, beta);
  ctx_kernel<<<(B * E + 255) / 256, 256, 0, stream>>>(aw, H, beta, ctx);

  // 8) gates = [emb[E_tm1], ctx]@W_ih^T + b_ih + h0@W_hh^T + b_hh
  build_x<<<(B * (Wd + E) + 255) / 256, 256, 0, stream>>>(E_tm1, emb, ctx, x);
  gemm_bias_act<<<gemm_grid(B, 4 * D), 128, 0, stream>>>(x, W_ih, b_ih, nullptr, gates,
      4 * D, Wd + E, Wd + E, Wd + E, 4 * D, FLAG_BIAS);
  gemm_bias_act<<<gemm_grid(B, 4 * D), 128, 0, stream>>>(h0, W_hh, b_hh, gates, gates,
      4 * D, D, D, D, 4 * D, FLAG_BIAS | FLAG_ADDC);

  // 9) LSTM pointwise -> h_t, c_t (outputs #2, #3)
  lstm_pointwise<<<(B * D + 255) / 256, 256, 0, stream>>>(gates, c0, h_t, c_t);

  // 10) logits = [h_t, emb[argmax(y_tm1)], ctx] @ W_out^T + b_out
  argmax_rows<<<B, 256, 0, stream>>>(y_tm1, amx);
  build_z<<<(B * (D + Wd + E) + 255) / 256, 256, 0, stream>>>(h_t, emb, amx, ctx, z);
  gemm_bias_act<<<gemm_grid(B, V), 128, 0, stream>>>(z, W_out, b_out, nullptr, logits,
      V, D + Wd + E, D + Wd + E, D + Wd + E, V, FLAG_BIAS);
}
